// ColorNet_38963943309479
// MI455X (gfx1250) — matrix operand order repaired and checked by reading, not compiled, not run on hardware
//
#include <hip/hip_runtime.h>
#include <hip/hip_bf16.h>

// ---------------------------------------------------------------------------
// MI455X (gfx1250) implementation.
//
//  K1: fused 7x7 box conv (s2,p3) + 3x3 maxpool (s2,p1): 512^2 -> 128^2.
//      Separable box filter via LDS, one HBM pass over the 134 MB input.
//  K2: one workgroup per image; entire 128x128 fp32 plane resident in LDS
//      (CDNA5: 320 KB/WGP). TDM async tile load (padded to a 130-f32 row
//      pitch, data at column origin 1) -> 27 LDS-resident 3x3 box convs as
//      banded-matrix WMMAs (v_wmma_f32_16x16x32_f16; f32 master plane, f16
//      staging; zero-padded LDS borders make all operand loads branchless,
//      and the bands' zeros at K>=18 make the K-pad lanes dead) -> TDM store
//      of the final 32x32 plane.
// ---------------------------------------------------------------------------

typedef __attribute__((ext_vector_type(16))) _Float16 v16h;
typedef __attribute__((ext_vector_type(8)))  _Float16 v8h;
typedef __attribute__((ext_vector_type(8)))  float    v8f;
typedef __attribute__((ext_vector_type(4)))  unsigned v4u;
typedef __attribute__((ext_vector_type(8)))  int      v8i;
typedef __attribute__((ext_vector_type(4)))  int      v4i;

// ============================ K1: front end ================================
// Pooled-output tile 16x16 per block. Input halo region: 71x71.
#define TIN    71
#define TIN_S  72
#define NH     33
#define NH_S   36

__global__ __launch_bounds__(256)
void frontend_conv7_pool(const float* __restrict__ x, float* __restrict__ out) {
    __shared__ float sIn[TIN * TIN_S];   // 20448 B
    __shared__ float sHs[TIN * NH_S];    // 10224 B

    const int img = blockIdx.z;
    const int OX  = blockIdx.x << 4;
    const int OY  = blockIdx.y << 4;
    const int tid = threadIdx.x;
    const float* src = x + (size_t)img * 512 * 512;

    const int iy0 = 4 * OY - 5;
    const int ix0 = 4 * OX - 5;

    // Speculative prefetch a few rows ahead (gfx1250 global_prefetch_b8).
    __builtin_prefetch(src + (size_t)((iy0 + TIN) & 511) * 512 + (ix0 & 511), 0, 1);

    // Cooperative zero-padded load of the 71x71 input halo tile.
    for (int i = tid; i < TIN * TIN; i += 256) {
        const int r = i / TIN, c = i % TIN;          // constant divisor: mul+shift
        const int gy = iy0 + r, gx = ix0 + c;
        float v = 0.0f;
        if ((unsigned)gy < 512u && (unsigned)gx < 512u) v = src[(size_t)gy * 512 + gx];
        sIn[r * TIN_S + c] = v;
    }
    __syncthreads();

    // Horizontal 7-tap sums at the 33 stride-2 conv sites, all 71 rows.
    for (int i = tid; i < TIN * NH; i += 256) {
        const int r = i / NH, ci = i % NH;
        const float* p = &sIn[r * TIN_S + 2 * ci];
        sHs[r * NH_S + ci] = p[0] + p[1] + p[2] + p[3] + p[4] + p[5] + p[6];
    }
    __syncthreads();

    // Vertical 7-tap + 3x3 max (pool pad uses -inf semantics: mask OOB conv sites).
    const int ty = tid >> 4, tx = tid & 15;
    float m = -3.402823466e38f;
    for (int dj = 0; dj < 3; ++dj) {
        const int j  = 2 * ty + dj;                 // local conv row   (0..32)
        const int cy = 2 * (OY + ty) - 1 + dj;      // global conv row
        for (int di = 0; di < 3; ++di) {
            const int i2 = 2 * tx + di;             // local conv col
            const int cx = 2 * (OX + tx) - 1 + di;  // global conv col
            if ((unsigned)cy >= 256u || (unsigned)cx >= 256u) continue;
            float s = 0.0f;
            for (int k = 0; k < 7; ++k) s += sHs[(2 * j + k) * NH_S + i2];
            m = fmaxf(m, s * (1.0f / 49.0f));
        }
    }
    out[((size_t)img * 128 + (OY + ty)) * 128 + (OX + tx)] = m;
}

// ====================== K2: 27-layer LDS-resident chain ====================
// sPlane: f32, pitch 130, data at cols 1..S, cols 0 and S+1 are zero pads.
// sH:     f16, pitch 136 (272 B row => 16-B aligned sites), data at rows
//         1..S, rows 0 and S+1 are zero pads.
#define LSTRIDE 130
#define HSTRIDE 136
#define NWAVES  16

__global__ __launch_bounds__(512)
void chain27_wmma(const float* __restrict__ src, float* __restrict__ dst) {
    __shared__ float    sPlane[LSTRIDE * 128];   // 66560 B
    __shared__ _Float16 sH[HSTRIDE * 130];       // 35360 B

    const int img  = blockIdx.x;
    const int tid  = threadIdx.x;
    const int lane = tid & 31;
    const int wave = tid >> 5;

    // ---- TDM async load: 128x128 fp32 -> LDS at col origin 1, pitch 130 ----
#if __has_builtin(__builtin_amdgcn_tensor_load_to_lds)
    if (wave == 0) {
        const unsigned long long ga =
            (unsigned long long)(src + (size_t)img * 128 * 128);
        // D# group 0: count=1, lds_addr (origin col 1), global_addr, type=2
        v4u g0 = { 1u,
                   (unsigned)(unsigned long long)(sPlane + 1),
                   (unsigned)(ga & 0xFFFFFFFFu),
                   (unsigned)((ga >> 32) & 0x01FFFFFFu) | (2u << 30) };
        // D# group 1: data_size=4B(2), pad_enable, pad_interval=128 DW (6),
        // pad_amount=2 DW (1); tensor 128x128, tile 128x128, dim0 stride 128.
        v8i g1 = { (2 << 16) | (1 << 20) | (6 << 22) | (1 << 25),
                   (128 << 16),   // tensor_dim0[15:0]  @ bits[63:48]
                   (128 << 16),   // tensor_dim1[15:0]  @ bits[95:80]
                   (128 << 16),   // tile_dim0          @ bits[127:112]
                   128,           // tile_dim1          @ bits[143:128]
                   128,           // tensor_dim0_stride @ bits[191:160]
                   0, 0 };
        v4i z4 = { 0, 0, 0, 0 };
#if __clang_major__ >= 23
        v8i z8 = { 0, 0, 0, 0, 0, 0, 0, 0 };
        __builtin_amdgcn_tensor_load_to_lds(g0, g1, z4, z4, z8, 0);
#else
        __builtin_amdgcn_tensor_load_to_lds(g0, g1, z4, z4, 0);
#endif
    }
#else
    for (int i = tid; i < 128 * 128; i += 512)
        sPlane[(i >> 7) * LSTRIDE + 1 + (i & 127)] = src[(size_t)img * 16384 + i];
#endif

    // Zero the pad column pair and pad rows (overlaps with the TDM transfer).
    for (int i = tid; i < 128; i += 512) {
        sPlane[i * LSTRIDE]       = 0.0f;    // col -1
        sPlane[i * LSTRIDE + 129] = 0.0f;    // col S
    }
    for (int i = tid; i < HSTRIDE; i += 512) {
        sH[i]                 = (_Float16)0.0f;   // row -1
        sH[129 * HSTRIDE + i] = (_Float16)0.0f;   // row S
    }
#if __has_builtin(__builtin_amdgcn_tensor_load_to_lds)
    if (wave == 0) __builtin_amdgcn_s_wait_tensorcnt(0);
#endif
    __syncthreads();

    // A-operand K placement for 16-bit 16x32 A (ISA 7.12.2):
    // lanes 0-15: K in {0..7, 16..23}; lanes 16-31: K in {8..15, 24..31}.
    const int M    = lane & 15;
    const int kb0  = (lane < 16) ? 0 : 8;
    const int kb1  = (lane < 16) ? 16 : 24;
    const int moff = (lane >> 4) << 3;   // D-matrix M offset (+8 for hi lanes)

    // Loop-invariant band operands. Both are zero for K >= 18, which makes
    // the data operands' K-pad elements (18..31) algebraically dead.
    v16h bBand;   // B[k][n] = 1/9 for k in {n, n+1, n+2}; N = lane & 15, K per table order
    v16h aBand;   // A[m][k] = 1   for k in {m, m+1, m+2}
    for (int j = 0; j < 16; ++j) {
        const int K = (j < 8) ? (kb0 + j) : (kb1 + (j - 8));
        bBand[j] = (K >= M && K <= M + 2) ? (_Float16)(1.0f / 9.0f)
                                          : (_Float16)0.0f;
        aBand[j] = (K >= M && K <= M + 2) ? (_Float16)1.0f : (_Float16)0.0f;
    }

    int S = 128, lgnt = 3;                       // nt = S/16 = 1<<lgnt
    for (int l = 0; l < 27; ++l) {
        const int stride2 = (l == 6 || l == 15);
        const int nt      = 1 << lgnt;
        const int ntiles  = nt * nt;

        // ---- Phase 1: horizontal. D = A(data, halo in K-pad) x B(1/9 band)
        // Branchless: 5 unconditional float2 LDS loads per lane; pad columns
        // supply the conv zeros; K>=18 slots are dead (band is zero there).
        for (int t = wave; t < ntiles; t += NWAVES) {
            const int y0 = (t >> lgnt) << 4;
            const int x0 = (t & (nt - 1)) << 4;

            const int rbase = (y0 + M) * LSTRIDE + x0;   // col -1 site (origin 1)
            const float2 p0 = *(const float2*)&sPlane[rbase + kb0 + 0];
            const float2 p1 = *(const float2*)&sPlane[rbase + kb0 + 2];
            const float2 p2 = *(const float2*)&sPlane[rbase + kb0 + 4];
            const float2 p3 = *(const float2*)&sPlane[rbase + kb0 + 6];
            const float2 p4 = *(const float2*)&sPlane[rbase + 16];  // K=16,17 (lo lanes)

            v16h a;
            a[0] = (_Float16)p0.x;  a[1] = (_Float16)p0.y;
            a[2] = (_Float16)p1.x;  a[3] = (_Float16)p1.y;
            a[4] = (_Float16)p2.x;  a[5] = (_Float16)p2.y;
            a[6] = (_Float16)p3.x;  a[7] = (_Float16)p3.y;
            a[8] = (_Float16)p4.x;  a[9] = (_Float16)p4.y;  // dead on hi lanes
            a[10] = (_Float16)0.0f; a[11] = (_Float16)0.0f;
            a[12] = (_Float16)0.0f; a[13] = (_Float16)0.0f;
            a[14] = (_Float16)0.0f; a[15] = (_Float16)0.0f;

            v8f c = {};
            v8f h = __builtin_amdgcn_wmma_f32_16x16x32_f16(
                        false, a, false, bBand, (short)0, c, false, false);

            const int sb = (y0 + 1 + moff) * HSTRIDE + x0 + (lane & 15);
            for (int r = 0; r < 8; ++r)
                sH[sb + r * HSTRIDE] = (_Float16)h[r];
        }
        __syncthreads();

        // ---- Phase 2: vertical. D = A(unit band) x B(h data, halo in K-pad)
        // Branchless: clamp row into the zero-pad row (legit for K=17 halo;
        // elements with K>=18 are dead anyway). Per element s: row y0+K(s)
        // (table-order K), column x0 + (lane & 15).
        for (int t = wave; t < ntiles; t += NWAVES) {
            const int y0 = (t >> lgnt) << 4;
            const int x0 = (t & (nt - 1)) << 4;

            v16h b;
            for (int s = 0; s < 16; ++s) {
                const int K = (s < 8) ? (kb0 + s) : (kb1 + (s - 8));
                int rr = y0 + K;                      // staged row (origin 1)
                if (rr > S + 1) rr = S + 1;           // zero pad / dead elements
                b[s] = sH[rr * HSTRIDE + x0 + M];
            }

            v8f c = {};
            v8f d = __builtin_amdgcn_wmma_f32_16x16x32_f16(
                        false, aBand, false, b, (short)0, c, false, false);

            const int xn = x0 + (lane & 15);
            if (!stride2) {
                const int pb = (y0 + moff) * LSTRIDE + 1 + xn;
                for (int r = 0; r < 8; ++r)
                    sPlane[pb + r * LSTRIDE] = d[r];
            } else if ((xn & 1) == 0) {
                // y parity == r parity (y0, moff even): keep even r only.
                const int pb = ((y0 + moff) >> 1) * LSTRIDE + 1 + (xn >> 1);
                for (int r = 0; r < 8; r += 2)
                    sPlane[pb + (r >> 1) * LSTRIDE] = d[r];
            }
        }
        __syncthreads();

        if (stride2) {
            S >>= 1; lgnt -= 1;
            // Re-zero the new boundary pads (col S, staging row S).
            for (int i = tid; i < S; i += 512)
                sPlane[i * LSTRIDE + 1 + S] = 0.0f;
            for (int i = tid; i < HSTRIDE; i += 512)
                sH[(S + 1) * HSTRIDE + i] = (_Float16)0.0f;
            __syncthreads();
        }
    }

    // ---- Compact the final 32x32 plane (reuse staging memory), TDM store ----
    float* outbuf = (float*)sH;
    for (int i = tid; i < 32 * 32; i += 512)
        outbuf[i] = sPlane[(i >> 5) * LSTRIDE + 1 + (i & 31)];
    __syncthreads();

#if __has_builtin(__builtin_amdgcn_tensor_store_from_lds)
    if (wave == 0) {
        const unsigned long long ga =
            (unsigned long long)(dst + (size_t)img * 1024);
        v4u g0 = { 1u,
                   (unsigned)(unsigned long long)outbuf,
                   (unsigned)(ga & 0xFFFFFFFFu),
                   (unsigned)((ga >> 32) & 0x01FFFFFFu) | (2u << 30) };
        v8i g1 = { (2 << 16),     // data_size=4B, no pad, no multicast
                   (32 << 16),    // tensor_dim0[15:0]  @ bits[63:48]
                   (32 << 16),    // tensor_dim1[15:0]  @ bits[95:80]
                   (32 << 16),    // tile_dim0          @ bits[127:112]
                   32,            // tile_dim1          @ bits[143:128]
                   32,            // tensor_dim0_stride @ bits[191:160]
                   0, 0 };
        v4i z4 = { 0, 0, 0, 0 };
#if __clang_major__ >= 23
        v8i z8 = { 0, 0, 0, 0, 0, 0, 0, 0 };
        __builtin_amdgcn_tensor_store_from_lds(g0, g1, z4, z4, z8, 0);
#else
        __builtin_amdgcn_tensor_store_from_lds(g0, g1, z4, z4, 0);
#endif
        __builtin_amdgcn_s_wait_tensorcnt(0);
    }
#else
    for (int i = tid; i < 32 * 32; i += 512)
        dst[(size_t)img * 1024 + i] = outbuf[i];
#endif
}

// ================================ launch ===================================
extern "C" void kernel_launch(void* const* d_in, const int* in_sizes, int n_in,
                              void* d_out, int out_size, void* d_ws, size_t ws_size,
                              hipStream_t stream) {
    const float* x   = (const float*)d_in[0];   // (128,1,512,512) fp32
    float*       mid = (float*)d_ws;            // (128,1,128,128) fp32 = 8.4 MB
    float*       out = (float*)d_out;           // (128,1,32,32)  fp32

    dim3 g1(8, 8, 128);
    frontend_conv7_pool<<<g1, dim3(256), 0, stream>>>(x, mid);
    chain27_wmma<<<dim3(128), dim3(512), 0, stream>>>(mid, out);
}
